// MHA_90744069030316
// MI455X (gfx1250) — matrix operand order repaired and checked by reading, not compiled, not run on hardware
//
#include <hip/hip_runtime.h>
#include <hip/hip_bf16.h>

// ---------------- WMMA types / helpers (gfx1250, wave32) ----------------
typedef __attribute__((ext_vector_type(16))) __bf16 v16bf;
typedef __attribute__((ext_vector_type(8)))  float  v8f;

union FragAB {
    v16bf v;
    unsigned int u[8];
};

__device__ __forceinline__ v8f wmma_bf16(const FragAB& a, const FragAB& b, v8f c) {
    return __builtin_amdgcn_wmma_f32_16x16x32_bf16(
        /*neg_a=*/false, a.v, /*neg_b=*/false, b.v,
        /*c_mod=*/(short)0, c, /*reuse_a=*/false, /*reuse_b=*/false);
}

// float -> bf16 (round to nearest even), stored as raw u16
__device__ __forceinline__ unsigned short f2bf(float f) {
    unsigned int u = __builtin_bit_cast(unsigned int, f);
    u += 0x7FFFu + ((u >> 16) & 1u);
    return (unsigned short)(u >> 16);
}

// ---- CDNA5 async global->LDS copy (16B per lane), compile-safe fallback ----
// Builtin signature (from hipcc diagnostic): (v4i AS1*, v4i AS3*, imm offset, imm cpol)
typedef int v4i __attribute__((vector_size(16)));
typedef __attribute__((address_space(1))) v4i av4i_g;
typedef __attribute__((address_space(3))) v4i av4i_l;

__device__ __forceinline__ void async_copy_b128(const void* gsrc, void* ldst) {
#if __has_builtin(__builtin_amdgcn_global_load_async_to_lds_b128)
    __builtin_amdgcn_global_load_async_to_lds_b128((av4i_g*)gsrc, (av4i_l*)ldst, 0, 0);
#else
    *(uint4*)ldst = *(const uint4*)gsrc;
#endif
}
__device__ __forceinline__ void async_wait0() {
#if __has_builtin(__builtin_amdgcn_s_wait_asynccnt)
    __builtin_amdgcn_s_wait_asynccnt(0);
#endif
}

// A-matrix 16x32 bf16 fragment from row-major u16 buffer (ISA 7.12.2 layout):
// lane m=L%16, kb=8*(L/16): u[0..3]=K kb..kb+7 ; u[4..7]=K kb+16..kb+23
__device__ __forceinline__ FragAB load_a_frag(const unsigned short* base, int row0, int ld, int lane) {
    FragAB f;
    const int m  = lane & 15;
    const int kb = (lane >> 4) << 3;
    const unsigned short* p = base + (row0 + m) * ld + kb;
#pragma unroll
    for (int j = 0; j < 4; ++j) {
        f.u[j]     = *(const unsigned int*)(p + 2 * j);
        f.u[4 + j] = *(const unsigned int*)(p + 16 + 2 * j);
    }
    return f;
}

// B-matrix 32x16 bf16 fragment; buffer holds B^T row-major (row = output col n):
// lane n=L%16, kb=8*(L/16): u[0..3]=K kb..kb+7 ; u[4..7]=K kb+16..kb+23
__device__ __forceinline__ FragAB load_b_frag(const unsigned short* base, int row0, int ld, int lane) {
    FragAB f;
    const int n  = lane & 15;
    const int kb = (lane >> 4) << 3;
    const unsigned short* p = base + (row0 + n) * ld + kb;
#pragma unroll
    for (int j = 0; j < 4; ++j) {
        f.u[j]     = *(const unsigned int*)(p + 2 * j);
        f.u[4 + j] = *(const unsigned int*)(p + 16 + 2 * j);
    }
    return f;
}

// ---------------- prep: layout + precision conversion ----------------
__global__ void cvt_x_kernel(const float* __restrict__ x, unsigned short* __restrict__ xt) {
    const int i = blockIdx.x * 256 + threadIdx.x;      // flat over input (coalesced reads)
    const int s = i & 1023;
    const int c = (i >> 10) & 511;
    const int n = i >> 19;
    xt[((n << 10) + s) * 512 + c] = f2bf(x[i]);
}

__global__ void cvt_w_kernel(const float* __restrict__ w, unsigned short* __restrict__ o, int nelem) {
    const int i = blockIdx.x * 256 + threadIdx.x;
    if (i < nelem) o[i] = f2bf(w[i]);
}

// ---------------- QKV GEMM: [8192,512] x [512,1536] -> q/k [nh][1024][64], v^T [nh][64][1024] ----
// block tile 256(M) x 64(N), K-step 32, double-buffered async LDS; 8 waves each 32x64.
__global__ __launch_bounds__(256)
void qkv_gemm_kernel(const unsigned short* __restrict__ xt,
                     const unsigned short* __restrict__ wq,
                     unsigned short* __restrict__ qb,
                     unsigned short* __restrict__ kb,
                     unsigned short* __restrict__ vtb) {
    __shared__ __align__(16) unsigned short As[2][256 * 40];
    __shared__ __align__(16) unsigned short Bs[2][64 * 40];
    const int tid  = threadIdx.x;
    const int lane = tid & 31;
    const int wave = tid >> 5;
    const int gm0  = blockIdx.y * 256;
    const int gn0  = blockIdx.x * 64;

    v8f acc[2][4];
#pragma unroll
    for (int mi = 0; mi < 2; ++mi)
#pragma unroll
        for (int t = 0; t < 4; ++t)
#pragma unroll
            for (int i = 0; i < 8; ++i) acc[mi][t][i] = 0.0f;

    auto issue_tile = [&](int k0, int buf) {
        // A tile: 256 rows x 32 cols, one row per thread (4 x b128)
        const unsigned short* ga = xt + (gm0 + tid) * 512 + k0;
        unsigned short* sa = &As[buf][tid * 40];
#pragma unroll
        for (int j = 0; j < 4; ++j) async_copy_b128(ga + j * 8, sa + j * 8);
        // B tile: 64 x 32 (rows of w_qkv are already B^T layout)
        const int r  = tid >> 2;
        const int c0 = (tid & 3) * 8;
        async_copy_b128(wq + (gn0 + r) * 512 + k0 + c0, &Bs[buf][r * 40 + c0]);
    };

    issue_tile(0, 0);
    async_wait0();
    __syncthreads();

    for (int kidx = 0; kidx < 16; ++kidx) {
        const int cur = kidx & 1;
        if (kidx + 1 < 16) issue_tile((kidx + 1) * 32, cur ^ 1);

        const FragAB a0 = load_a_frag(As[cur], wave * 32, 40, lane);
        const FragAB a1 = load_a_frag(As[cur], wave * 32 + 16, 40, lane);
        FragAB bf[4];
#pragma unroll
        for (int t = 0; t < 4; ++t) bf[t] = load_b_frag(Bs[cur], t * 16, 40, lane);
#pragma unroll
        for (int t = 0; t < 4; ++t) {
            acc[0][t] = wmma_bf16(a0, bf[t], acc[0][t]);
            acc[1][t] = wmma_bf16(a1, bf[t], acc[1][t]);
        }
        if (kidx + 1 < 16) { async_wait0(); __syncthreads(); }
    }

    // epilogue: q/k -> [nh][s][d], v -> transposed [nh][d][s] (for async attention staging)
#pragma unroll
    for (int mi = 0; mi < 2; ++mi) {
        const int mrow0 = gm0 + wave * 32 + mi * 16 + ((lane >> 4) << 3);
#pragma unroll
        for (int t = 0; t < 4; ++t) {
            const int e     = gn0 + t * 16 + (lane & 15);
            const int which = e >> 9;          // uniform per block
            const int rem   = e & 511;
            const int head  = rem >> 6;
            const int d     = rem & 63;
            if (which == 0) {
#pragma unroll
                for (int r = 0; r < 8; ++r) {
                    const int sf = mrow0 + r;
                    qb[(((sf >> 10 << 3) + head) << 16) + ((sf & 1023) << 6) + d] =
                        f2bf(acc[mi][t][r] * 0.125f);   // 1/sqrt(hd) fused
                }
            } else if (which == 1) {
#pragma unroll
                for (int r = 0; r < 8; ++r) {
                    const int sf = mrow0 + r;
                    kb[(((sf >> 10 << 3) + head) << 16) + ((sf & 1023) << 6) + d] =
                        f2bf(acc[mi][t][r]);
                }
            } else {
#pragma unroll
                for (int r = 0; r < 8; ++r) {
                    const int sf = mrow0 + r;
                    vtb[(((sf >> 10 << 3) + head) << 16) + (d << 10) + (sf & 1023)] =
                        f2bf(acc[mi][t][r]);
                }
            }
        }
    }
}

// ---------------- flash attention: per (n,h), 8 waves x 16-query tiles ----------------
// K and V^T tiles double-buffered via async global->LDS.
__global__ __launch_bounds__(256)
void attn_kernel(const unsigned short* __restrict__ qb,
                 const unsigned short* __restrict__ kbuf,
                 const unsigned short* __restrict__ vtbuf,
                 unsigned short* __restrict__ ao) {
    __shared__ __align__(16) unsigned short Ks[2][32 * 72];    // [key][dim]  ld 72
    __shared__ __align__(16) unsigned short Vts[2][64 * 40];   // [dim][key]  ld 40
    __shared__ __align__(16) unsigned short Ps[8][16 * 40];    // per-wave P, ld 40

    const int tid  = threadIdx.x;
    const int lane = tid & 31;
    const int wave = tid >> 5;
    const int nh   = blockIdx.y;
    const int q0   = (blockIdx.x * 8 + wave) * 16;

    const unsigned short* Q  = qb    + (nh << 16);
    const unsigned short* K  = kbuf  + (nh << 16);
    const unsigned short* Vt = vtbuf + (nh << 16);

    // Q fragments (held in registers for the whole key loop)
    FragAB aq[2];
    {
        const int m  = lane & 15;
        const int kbq = (lane >> 4) << 3;
        const unsigned short* qp = Q + (q0 + m) * 64;
#pragma unroll
        for (int h = 0; h < 2; ++h) {
#pragma unroll
            for (int j = 0; j < 4; ++j) {
                aq[h].u[j]     = *(const unsigned int*)(qp + h * 32 + kbq + 2 * j);
                aq[h].u[4 + j] = *(const unsigned int*)(qp + h * 32 + kbq + 16 + 2 * j);
            }
        }
    }

    v8f   o[4];
    float mrow[8], lrow[8];
#pragma unroll
    for (int t = 0; t < 4; ++t)
#pragma unroll
        for (int i = 0; i < 8; ++i) o[t][i] = 0.0f;
#pragma unroll
    for (int r = 0; r < 8; ++r) { mrow[r] = -1.0e30f; lrow[r] = 0.0f; }

    unsigned short* myPs = Ps[wave];

    auto issue_tile = [&](int ks0, int buf) {
        {   // K tile: 32 keys x 64 dims
            const int r  = tid >> 3;
            const int c0 = (tid & 7) * 8;
            async_copy_b128(K + (ks0 + r) * 64 + c0, &Ks[buf][r * 72 + c0]);
        }
        {   // V^T tile: 64 dims x 32 keys (already transposed in global)
            const int r  = tid >> 2;
            const int c0 = (tid & 3) * 8;
            async_copy_b128(Vt + (r << 10) + ks0 + c0, &Vts[buf][r * 40 + c0]);
        }
    };

    issue_tile(0, 0);
    async_wait0();
    __syncthreads();

    for (int j = 0; j < 32; ++j) {
        const int cur = j & 1;
        const int ks0 = j * 32;
        if (j + 1 < 32) issue_tile(ks0 + 32, cur ^ 1);

        // scores: 16 queries x 32 keys, K-dim = hd = 64 -> 4 WMMAs
        v8f sc[2];
#pragma unroll
        for (int i = 0; i < 8; ++i) { sc[0][i] = 0.0f; sc[1][i] = 0.0f; }
        {
            FragAB bk[2][2];
#pragma unroll
            for (int hh = 0; hh < 2; ++hh)
#pragma unroll
                for (int sub = 0; sub < 2; ++sub)
                    bk[hh][sub] = load_b_frag(&Ks[cur][hh * 32], sub * 16, 72, lane);
#pragma unroll
            for (int hh = 0; hh < 2; ++hh)
#pragma unroll
                for (int sub = 0; sub < 2; ++sub)
                    sc[sub] = wmma_bf16(aq[hh], bk[hh][sub], sc[sub]);
        }

        // online softmax (rows live per 16-lane half -> xor reduce with masks 8..1)
#pragma unroll
        for (int r = 0; r < 8; ++r) {
            const float x0 = sc[0][r], x1 = sc[1][r];
            float mx = fmaxf(x0, x1);
#pragma unroll
            for (int off = 8; off >= 1; off >>= 1) mx = fmaxf(mx, __shfl_xor(mx, off, 32));
            const float newm = fmaxf(mrow[r], mx);
            const float corr = __expf(mrow[r] - newm);
            const float p0   = __expf(x0 - newm);
            const float p1   = __expf(x1 - newm);
            float rs = p0 + p1;
#pragma unroll
            for (int off = 8; off >= 1; off >>= 1) rs += __shfl_xor(rs, off, 32);
            lrow[r] = lrow[r] * corr + rs;
            mrow[r] = newm;
#pragma unroll
            for (int t = 0; t < 4; ++t) o[t][r] *= corr;
            const int m = r + ((lane >> 4) << 3);
            const int n = lane & 15;
            myPs[m * 40 + n]      = f2bf(p0);
            myPs[m * 40 + n + 16] = f2bf(p1);
        }

        // P @ V  (same-wave LDS is in-order: no barrier needed for myPs)
        {
            const FragAB ap = load_a_frag(myPs, 0, 40, lane);
            FragAB bv[4];
#pragma unroll
            for (int t = 0; t < 4; ++t) bv[t] = load_b_frag(Vts[cur], t * 16, 40, lane);
#pragma unroll
            for (int t = 0; t < 4; ++t) o[t] = wmma_bf16(ap, bv[t], o[t]);
        }

        if (j + 1 < 32) { async_wait0(); __syncthreads(); }
    }

    // normalize + store to attn_out [n][s][c] bf16
#pragma unroll
    for (int r = 0; r < 8; ++r) lrow[r] = 1.0f / lrow[r];
    const int n_img = nh >> 3, head = nh & 7;
#pragma unroll
    for (int t = 0; t < 4; ++t) {
        const int c = (head << 6) + t * 16 + (lane & 15);
#pragma unroll
        for (int r = 0; r < 8; ++r) {
            const int s = q0 + r + ((lane >> 4) << 3);
            ao[((n_img << 10) + s) * 512 + c] = f2bf(o[t][r] * lrow[r]);
        }
    }
}

// ---------------- output projection: [8192,512] x [512,512] + bias -> NCHW f32 ----------------
__global__ __launch_bounds__(256)
void outproj_kernel(const unsigned short* __restrict__ A,
                    const unsigned short* __restrict__ W,
                    const float* __restrict__ bias,
                    float* __restrict__ out) {
    __shared__ __align__(16) unsigned short As[2][256 * 40];
    __shared__ __align__(16) unsigned short Bs[2][64 * 40];
    const int tid  = threadIdx.x;
    const int lane = tid & 31;
    const int wave = tid >> 5;
    const int gm0  = blockIdx.y * 256;
    const int gn0  = blockIdx.x * 64;

    v8f acc[2][4];
#pragma unroll
    for (int mi = 0; mi < 2; ++mi)
#pragma unroll
        for (int t = 0; t < 4; ++t)
#pragma unroll
            for (int i = 0; i < 8; ++i) acc[mi][t][i] = 0.0f;

    auto issue_tile = [&](int k0, int buf) {
        const unsigned short* ga = A + (gm0 + tid) * 512 + k0;
        unsigned short* sa = &As[buf][tid * 40];
#pragma unroll
        for (int j = 0; j < 4; ++j) async_copy_b128(ga + j * 8, sa + j * 8);
        const int r  = tid >> 2;
        const int c0 = (tid & 3) * 8;
        async_copy_b128(W + (gn0 + r) * 512 + k0 + c0, &Bs[buf][r * 40 + c0]);
    };

    issue_tile(0, 0);
    async_wait0();
    __syncthreads();

    for (int kidx = 0; kidx < 16; ++kidx) {
        const int cur = kidx & 1;
        if (kidx + 1 < 16) issue_tile((kidx + 1) * 32, cur ^ 1);

        const FragAB a0 = load_a_frag(As[cur], wave * 32, 40, lane);
        const FragAB a1 = load_a_frag(As[cur], wave * 32 + 16, 40, lane);
        FragAB bf[4];
#pragma unroll
        for (int t = 0; t < 4; ++t) bf[t] = load_b_frag(Bs[cur], t * 16, 40, lane);
#pragma unroll
        for (int t = 0; t < 4; ++t) {
            acc[0][t] = wmma_bf16(a0, bf[t], acc[0][t]);
            acc[1][t] = wmma_bf16(a1, bf[t], acc[1][t]);
        }
        if (kidx + 1 < 16) { async_wait0(); __syncthreads(); }
    }

    // epilogue: lane's 8 rows contiguous in s -> two b128 stores per fragment
#pragma unroll
    for (int mi = 0; mi < 2; ++mi) {
        const int mrow0 = gm0 + wave * 32 + mi * 16 + ((lane >> 4) << 3);
        const int n_img = mrow0 >> 10;
        const int s0    = mrow0 & 1023;
#pragma unroll
        for (int t = 0; t < 4; ++t) {
            const int oc  = gn0 + t * 16 + (lane & 15);
            const float b = bias[oc];
            float* dst = out + ((size_t)(n_img << 9) + oc) * 1024 + s0;
            const float4 lo = make_float4(acc[mi][t][0] + b, acc[mi][t][1] + b,
                                          acc[mi][t][2] + b, acc[mi][t][3] + b);
            const float4 hi = make_float4(acc[mi][t][4] + b, acc[mi][t][5] + b,
                                          acc[mi][t][6] + b, acc[mi][t][7] + b);
            *(float4*)(dst)     = lo;
            *(float4*)(dst + 4) = hi;
        }
    }
}

// ---------------- launcher ----------------
extern "C" void kernel_launch(void* const* d_in, const int* in_sizes, int n_in,
                              void* d_out, int out_size, void* d_ws, size_t ws_size,
                              hipStream_t stream) {
    const float* x     = (const float*)d_in[0];
    const float* w_qkv = (const float*)d_in[1];
    const float* w_out = (const float*)d_in[2];
    const float* b_out = (const float*)d_in[3];
    float* out = (float*)d_out;

    // workspace carve-out (~42 MB), all chunks 256B-aligned
    char* ws = (char*)d_ws;
    unsigned short* xt  = (unsigned short*)ws;  ws += (size_t)8192 * 512 * 2;
    unsigned short* wq  = (unsigned short*)ws;  ws += (size_t)1536 * 512 * 2;
    unsigned short* wo  = (unsigned short*)ws;  ws += (size_t)512 * 512 * 2;
    unsigned short* qb  = (unsigned short*)ws;  ws += (size_t)64 * 1024 * 64 * 2;
    unsigned short* kb  = (unsigned short*)ws;  ws += (size_t)64 * 1024 * 64 * 2;
    unsigned short* vtb = (unsigned short*)ws;  ws += (size_t)64 * 64 * 1024 * 2;
    unsigned short* ao  = (unsigned short*)ws;  ws += (size_t)8192 * 512 * 2;

    cvt_x_kernel<<<16384, 256, 0, stream>>>(x, xt);
    cvt_w_kernel<<<3072, 256, 0, stream>>>(w_qkv, wq, 1536 * 512);
    cvt_w_kernel<<<1024, 256, 0, stream>>>(w_out, wo, 512 * 512);

    qkv_gemm_kernel<<<dim3(24, 32), 256, 0, stream>>>(xt, wq, qb, kb, vtb);
    attn_kernel<<<dim3(8, 64), 256, 0, stream>>>(qb, kb, vtb, ao);
    outproj_kernel<<<dim3(8, 32), 256, 0, stream>>>(ao, wo, b_out, out);
}
